// TwoStageModel_4346506904188
// MI455X (gfx1250) — compile-verified, matrix operand order checked by reading, not run on hardware
//
#include <hip/hip_runtime.h>
#include <hip/hip_bf16.h>

// ---------------------------------------------------------------------------
// TwoStageModel for MI455X (gfx1250): bf16 WMMA everywhere, f32 accumulate.
// Memory-bound on f32 weights (~440MB/call -> ~20us floor @ 23.3 TB/s), so
// weights are read once and converted f32->bf16 in flight (regs -> LDS).
// GEMM A-tiles (already bf16) are staged via the Tensor Data Mover (TDM)
// with LDS padding, fenced by s_wait_tensorcnt.
// ---------------------------------------------------------------------------

#define DEVINL __device__ __forceinline__

#if defined(__has_builtin)
#if __has_builtin(__builtin_amdgcn_tensor_load_to_lds) && \
    __has_builtin(__builtin_amdgcn_s_wait_tensorcnt)
#define USE_TDM 1
#endif
#endif

constexpr int B_  = 4;
constexpr int L_  = 512;
constexpr int D_  = 1024;
constexpr int H_  = 16;
constexpr int NL_ = 4;
constexpr int V_  = 8192;
constexpr int HD_ = D_ / H_;   // 64
constexpr int M_  = B_ * L_;   // 2048 tokens

typedef __attribute__((ext_vector_type(16))) __bf16 v16bf;
typedef __attribute__((ext_vector_type(8)))  float  v8f;
typedef __attribute__((ext_vector_type(4)))  unsigned u32x4;
typedef __attribute__((ext_vector_type(8)))  int      i32x8;
typedef __attribute__((ext_vector_type(4)))  int      i32x4;

union FragA { unsigned u[8]; v16bf v; };

DEVINL unsigned short bfb(float f) {            // f32 -> bf16 (RNE)
  unsigned u = __float_as_uint(f);
  u += 0x7FFFu + ((u >> 16) & 1u);
  return (unsigned short)(u >> 16);
}
DEVINL unsigned pack2(float a, float b) {
  return (unsigned)bfb(a) | ((unsigned)bfb(b) << 16);
}
// VGPR v of a 16-bit A/B fragment holds K-pair starting at this k:
DEVINL int kmap(int v, int half) { return ((v < 4) ? 2 * v : 16 + 2 * (v - 4)) + 8 * half; }

DEVINL v8f wmma_bf16(const FragA& a, const FragA& b, v8f c) {
  return __builtin_amdgcn_wmma_f32_16x16x32_bf16(
      /*neg_a=*/false, a.v, /*neg_b=*/false, b.v,
      /*c_mod=*/(short)0, c, /*reuse_a=*/false, /*reuse_b=*/false);
}

DEVINL float gelu_tanh(float x) {
  float x3 = x * x * x;
  return 0.5f * x * (1.0f + tanhf(0.7978845608028654f * (x + 0.044715f * x3)));
}

// ---------------------------------------------------------------------------
// Embedding kernels
// ---------------------------------------------------------------------------
__global__ __launch_bounds__(256)
void embed1_kernel(const int* __restrict__ ids, const int* __restrict__ smask,
                   const float* __restrict__ tok, const float* __restrict__ stok,
                   const float* __restrict__ pos, float* __restrict__ out) {
  const int t = blockIdx.x;
  const int l = t & (L_ - 1);
  const int id = ids[t] < 0 ? 0 : ids[t];
  const float* src = (smask[t] != 0) ? stok : (tok + (size_t)id * D_);
  const int d = threadIdx.x * 4;
  const float4 a = *(const float4*)(src + d);
  const float4 p = *(const float4*)(pos + (size_t)l * D_ + d);
  float4 r;
  r.x = a.x + p.x; r.y = a.y + p.y; r.z = a.z + p.z; r.w = a.w + p.w;
  *(float4*)(out + (size_t)t * D_ + d) = r;
}

__global__ __launch_bounds__(256)
void embed2_kernel(const int* __restrict__ ids, const int* __restrict__ smask,
                   const float* __restrict__ tok, const float* __restrict__ h1,
                   const float* __restrict__ pos, float* __restrict__ out) {
  const int t = blockIdx.x;
  const int l = t & (L_ - 1);
  const int id = ids[t] < 0 ? 0 : ids[t];
  const float* src = (smask[t] != 0) ? (h1 + (size_t)t * D_) : (tok + (size_t)id * D_);
  const int d = threadIdx.x * 4;
  const float4 a = *(const float4*)(src + d);
  const float4 p = *(const float4*)(pos + (size_t)l * D_ + d);
  float4 r;
  r.x = a.x + p.x; r.y = a.y + p.y; r.z = a.z + p.z; r.w = a.w + p.w;
  *(float4*)(out + (size_t)t * D_ + d) = r;
}

// last_s[b][i] = cummax over j<=i of (smask ? j : -1)
__global__ void lasts_kernel(const int* __restrict__ smask, int* __restrict__ lastS) {
  const int b = threadIdx.x;
  if (b >= B_) return;
  int last = -1;
  for (int i = 0; i < L_; ++i) {
    if (smask[b * L_ + i] != 0) last = i;
    lastS[b * L_ + i] = last;
  }
}

// ---------------------------------------------------------------------------
// LayerNorm: one 256-thread block per token row (D=1024 -> 4 f32/thread)
// ---------------------------------------------------------------------------
template <bool WB, bool WF>
__global__ __launch_bounds__(256)
void ln_kernel(const float* __restrict__ h, const float* __restrict__ g,
               const float* __restrict__ bt, unsigned short* __restrict__ outB,
               float* __restrict__ outF) {
  const int row = blockIdx.x;
  const int tid = threadIdx.x;
  const float4 v = *(const float4*)(h + (size_t)row * D_ + tid * 4);
  float s = v.x + v.y + v.z + v.w;
  float q = v.x * v.x + v.y * v.y + v.z * v.z + v.w * v.w;
#pragma unroll
  for (int m = 16; m >= 1; m >>= 1) {
    s += __shfl_xor(s, m, 32);
    q += __shfl_xor(q, m, 32);
  }
  __shared__ float ss[8], qq[8];
  if ((tid & 31) == 0) { ss[tid >> 5] = s; qq[tid >> 5] = q; }
  __syncthreads();
  float ts = 0.f, tq = 0.f;
#pragma unroll
  for (int i = 0; i < 8; ++i) { ts += ss[i]; tq += qq[i]; }
  const float mean = ts * (1.0f / D_);
  const float var  = tq * (1.0f / D_) - mean * mean;
  const float rstd = rsqrtf(var + 1e-5f);
  const float xs[4] = {v.x, v.y, v.z, v.w};
#pragma unroll
  for (int e = 0; e < 4; ++e) {
    const int d = tid * 4 + e;
    const float y = (xs[e] - mean) * rstd * g[d] + bt[d];
    if (WB) outB[(size_t)row * D_ + d] = bfb(y);
    if (WF) outF[(size_t)row * D_ + d] = y;
  }
}

// ---------------------------------------------------------------------------
// Tiled GEMM: out[M,N] = act(A_bf16[M,K] @ W_f32->bf16[K,N] + bias [+ resid])
// Block tile 128x128, K-step 32 (one WMMA K). 256 threads = 8 waves (2x4).
// Each wave: 64x32 region = 4x2 accumulators of 16x16 f32.
// A tile staged by TDM (per-wave 16-row 2D tile descriptor, LDS pad keeps the
// 40-elem padded stride); W tile staged via VALU f32->bf16 conversion.
// LDS strides padded (40 / 36 elems) so fragment ds_load_b32/b128 is
// bank-conflict-free.
// ---------------------------------------------------------------------------
constexpr int AST = 40;  // As row stride (bf16 elems); 80B rows -> 16B-aligned
constexpr int BST = 36;  // Bs row stride (bf16 elems); stored transposed [n][k]

template <bool RES, bool GELU, bool OUTBF>
__global__ __launch_bounds__(256)
void gemm_kernel(const unsigned short* __restrict__ A, const float* __restrict__ W,
                 const float* __restrict__ bias, const float* __restrict__ resid,
                 float* __restrict__ outF, unsigned short* __restrict__ outB,
                 int K, int N) {
  __shared__ unsigned short As[128 * AST];
  __shared__ unsigned short Bs[128 * BST];
  const int tid   = threadIdx.x;
  const int wid   = tid >> 5;
  const int lane  = tid & 31;
  const int half  = lane >> 4;
  const int l16   = lane & 15;
  const int waveM = wid >> 2;      // 0..1
  const int waveN = wid & 3;       // 0..3
  const int blockM = blockIdx.y * 128;
  const int blockN = blockIdx.x * 128;

  const v8f vz = {0.f, 0.f, 0.f, 0.f, 0.f, 0.f, 0.f, 0.f};
  v8f acc[4][2];
#pragma unroll
  for (int mt = 0; mt < 4; ++mt)
#pragma unroll
    for (int nt = 0; nt < 2; ++nt) acc[mt][nt] = vz;

  const int bn = tid & 127;        // W: column within tile
  const int bk = (tid >> 7) * 4;   // W: k group base (0 or 4)
#if !defined(USE_TDM)
  const int ac = tid & 3;          // A: 16B chunk within 32-elem row
  const int ar = tid >> 2;         // A: row 0..63 (x2 passes)
#endif

  for (int k0 = 0; k0 < K; k0 += 32) {
#if defined(USE_TDM)
    // --- stage A tile (128x32 bf16) via Tensor Data Mover ---------------
    // Per-wave 2D tile: 16 rows x 32 elems of the (M_,K) bf16 tensor, with
    // LDS padding 4 DWORDs after every 16 DWORDs (row) -> 40-elem stride.
    {
      const unsigned lds_off =
          (unsigned)(size_t)(&As[0]) + (unsigned)(wid * 16 * AST * 2);
      const unsigned long long ga =
          (unsigned long long)(size_t)(A + (size_t)(blockM + wid * 16) * K + k0);
      u32x4 g0;
      g0[0] = 1u;                                  // count=1 (valid descriptor)
      g0[1] = lds_off;                             // lds_addr (bytes)
      g0[2] = (unsigned)(ga & 0xFFFFFFFFull);      // global_addr[31:0]
      g0[3] = ((unsigned)(ga >> 32) & 0x01FFFFFFu) // global_addr[56:32]
              | (2u << 30);                        // type=2 ("image")
      i32x8 g1;
      g1[0] = (int)((1u << 16)                     // data_size = 2 bytes
                    | (1u << 20)                   // pad_enable
                    | (3u << 22)                   // pad_interval = 16 DWORDs
                    | (3u << 25));                 // pad_amount   = 4 DWORDs
      g1[1] = (int)(((unsigned)K & 0xFFFFu) << 16);        // tensor_dim0[15:0]
      g1[2] = (int)((((unsigned)K >> 16) & 0xFFFFu)        // tensor_dim0[31:16]
                    | (((unsigned)M_ & 0xFFFFu) << 16));   // tensor_dim1[15:0]
      g1[3] = (int)(32u << 16);                    // tensor_dim1 hi, tile_dim0=32
      g1[4] = 16;                                  // tile_dim1=16, tile_dim2=0
      g1[5] = K;                                   // tensor_dim0_stride[31:0]
      g1[6] = 0;                                   // stride hi / dim1_stride lo
      g1[7] = 0;
      const i32x4 gz4 = {0, 0, 0, 0};
      const i32x8 gz8 = {0, 0, 0, 0, 0, 0, 0, 0};
      // 6-arg form (clang-23 / therock-10.0 headers)
      __builtin_amdgcn_tensor_load_to_lds(g0, g1, gz4, gz4, gz8, 0);
    }
#else
    // --- fallback: stage A tile through VGPRs ---------------------------
#pragma unroll
    for (int p = 0; p < 2; ++p) {
      const int m = ar + p * 64;
      const uint4 va = *(const uint4*)(A + (size_t)(blockM + m) * K + k0 + ac * 8);
      *(uint4*)(&As[m * AST + ac * 8]) = va;
    }
#endif
    // --- stage W tile (32x128 f32 -> bf16, transposed to [n][k]) --------
#pragma unroll
    for (int p = 0; p < 4; ++p) {
      const int kk = bk + p * 8;   // {0,4}+{0,8,16,24}: all mult-of-4 in [0,28]
      const float* wp = W + (size_t)(k0 + kk) * N + blockN + bn;
      const float f0 = wp[0];
      const float f1 = wp[(size_t)N];
      const float f2 = wp[(size_t)2 * N];
      const float f3 = wp[(size_t)3 * N];
      uint2 pv;
      pv.x = pack2(f0, f1);
      pv.y = pack2(f2, f3);
      *(uint2*)(&Bs[bn * BST + kk]) = pv;
    }
    if (k0 + 32 < K) {  // hint next K-tile into the cache hierarchy
      __builtin_prefetch(A + (size_t)(blockM + (tid >> 2)) * K + k0 + 32, 0, 0);
      __builtin_prefetch(W + (size_t)(k0 + 32 + bk) * N + blockN + bn, 0, 0);
    }
#if defined(USE_TDM)
    __builtin_amdgcn_s_wait_tensorcnt(0);   // own wave's TDM tile landed
#endif
    __syncthreads();                        // everyone's tiles visible

    // --- fragments in ISA layout + 8 WMMAs ------------------------------
    FragA af[4], bf[2];
#pragma unroll
    for (int mt = 0; mt < 4; ++mt) {
      const int m = waveM * 64 + mt * 16 + l16;
#pragma unroll
      for (int v = 0; v < 8; ++v)
        af[mt].u[v] = *(const unsigned*)(&As[m * AST + kmap(v, half)]);
    }
#pragma unroll
    for (int nt = 0; nt < 2; ++nt) {
      const int n = waveN * 32 + nt * 16 + l16;
#pragma unroll
      for (int v = 0; v < 8; ++v)
        bf[nt].u[v] = *(const unsigned*)(&Bs[n * BST + kmap(v, half)]);
    }
#pragma unroll
    for (int mt = 0; mt < 4; ++mt)
#pragma unroll
      for (int nt = 0; nt < 2; ++nt)
        acc[mt][nt] = wmma_bf16(af[mt], bf[nt], acc[mt][nt]);
    __syncthreads();
  }

  // --- epilogue: C layout lane holds (row = r + 8*half, col = l16) ------
#pragma unroll
  for (int nt = 0; nt < 2; ++nt) {
    const int gn = blockN + waveN * 32 + nt * 16 + l16;
    const float bv = bias[gn];
#pragma unroll
    for (int mt = 0; mt < 4; ++mt) {
#pragma unroll
      for (int r = 0; r < 8; ++r) {
        const int gm = blockM + waveM * 64 + mt * 16 + r + 8 * half;
        float x = acc[mt][nt][r] + bv;
        if (RES)   x += resid[(size_t)gm * N + gn];
        if (GELU)  x  = gelu_tanh(x);
        if (OUTBF) outB[(size_t)gm * N + gn] = bfb(x);
        else       outF[(size_t)gm * N + gn] = x;
      }
    }
  }
}

// ---------------------------------------------------------------------------
// Flash attention: block = 4 waves x 16 query rows; j-blocks of 32 keys.
// S = QK^T (2 WMMA/tile), online softmax in VALU, P relayout C->A via
// per-wave LDS (same-wave DS ops are in-order), O += P@V (4 WMMA).
// qkv layout: (B*L, 3D) rows; q|k|v at col offsets 0|D|2D, head at h*HD.
// ---------------------------------------------------------------------------
template <bool SEG>
__global__ __launch_bounds__(128)
void attn_kernel(const float* __restrict__ qkv, const int* __restrict__ lastS,
                 unsigned short* __restrict__ outB) {
  const int bh = blockIdx.x;
  const int b  = bh / H_;
  const int hh = bh % H_;
  const int tid  = threadIdx.x;
  const int wid  = tid >> 5;
  const int lane = tid & 31;
  const int half = lane >> 4;
  const int l16  = lane & 15;
  const int i0   = blockIdx.y * 64 + wid * 16;

  __shared__ float PS[4][16 * 32];   // per-wave P staging (C-layout -> A-layout)
  float* ps = PS[wid];

  const float  scale = 0.125f;       // 1/sqrt(64)
  const size_t rs    = (size_t)3 * D_;
  const float* base  = qkv + (size_t)b * L_ * rs + hh * HD_;

  // Q fragments (A layout), features 0..31 and 32..63
  FragA qf[2];
#pragma unroll
  for (int f = 0; f < 2; ++f) {
    const float* qp = base + (size_t)(i0 + l16) * rs + f * 32;
#pragma unroll
    for (int v = 0; v < 8; ++v) {
      const int k = kmap(v, half);
      qf[f].u[v] = pack2(qp[k], qp[k + 1]);
    }
  }

  const v8f vz = {0.f, 0.f, 0.f, 0.f, 0.f, 0.f, 0.f, 0.f};
  v8f o[4];
#pragma unroll
  for (int t = 0; t < 4; ++t) o[t] = vz;
  float rmax[8], rsum[8];
#pragma unroll
  for (int r = 0; r < 8; ++r) { rmax[r] = -1e30f; rsum[r] = 0.f; }

  const int imax = i0 + 15;
  for (int j0 = 0; j0 <= imax; j0 += 32) {
    // ---- S = Q @ K^T for two 16-key sub-blocks ----
    v8f s[2];
#pragma unroll
    for (int sb = 0; sb < 2; ++sb) {
      FragA kf0, kf1;   // B layout: lane = key, VGPR = feature pair
      const float* kp = base + D_ + (size_t)(j0 + sb * 16 + l16) * rs;
#pragma unroll
      for (int v = 0; v < 8; ++v) {
        const int k = kmap(v, half);
        kf0.u[v] = pack2(kp[k], kp[k + 1]);
        kf1.u[v] = pack2(kp[32 + k], kp[32 + k + 1]);
      }
      s[sb] = wmma_bf16(qf[0], kf0, vz);
      s[sb] = wmma_bf16(qf[1], kf1, s[sb]);
    }

    // ---- mask + online softmax (row = r + 8*half, col = l16 per sub-block) ----
#pragma unroll
    for (int r = 0; r < 8; ++r) {
      const int i = i0 + r + 8 * half;
      int ls = 0;
      if (SEG) ls = lastS[b * L_ + i];
      float e0, e1;
      {
        const int j = j0 + l16;
        float v = s[0][r] * scale;
        if (j > i) v = -1e30f;
        if (SEG && !((j < i) && (j >= ls))) v += -1e9f;
        e0 = v;
      }
      {
        const int j = j0 + 16 + l16;
        float v = s[1][r] * scale;
        if (j > i) v = -1e30f;
        if (SEG && !((j < i) && (j >= ls))) v += -1e9f;
        e1 = v;
      }
      float mx = fmaxf(e0, e1);
#pragma unroll
      for (int m = 8; m >= 1; m >>= 1) mx = fmaxf(mx, __shfl_xor(mx, m, 32));
      const float nm   = fmaxf(rmax[r], mx);
      const float resc = __expf(rmax[r] - nm);
      const float p0   = __expf(e0 - nm);
      const float p1   = __expf(e1 - nm);
      float psum = p0 + p1;
#pragma unroll
      for (int m = 8; m >= 1; m >>= 1) psum += __shfl_xor(psum, m, 32);
      rsum[r] = rsum[r] * resc + psum;
      rmax[r] = nm;
#pragma unroll
      for (int t = 0; t < 4; ++t) o[t][r] *= resc;
      ps[(r + 8 * half) * 32 + l16]      = p0;
      ps[(r + 8 * half) * 32 + 16 + l16] = p1;
    }
    asm volatile("s_wait_dscnt 0" ::: "memory");  // same-wave DS store->load

    // ---- reload P as A-layout bf16 fragment (16x32) ----
    FragA pf;
#pragma unroll
    for (int v = 0; v < 8; ++v) {
      const int k = kmap(v, half);
      pf.u[v] = pack2(ps[l16 * 32 + k], ps[l16 * 32 + k + 1]);
    }

    // ---- O += P @ V : 4 feature tiles of 16 ----
    const float* vb = base + 2 * D_;
#pragma unroll
    for (int nt = 0; nt < 4; ++nt) {
      FragA vf;   // B layout: lane = feature col, VGPR = key pair
      const int n = nt * 16 + l16;
#pragma unroll
      for (int v = 0; v < 8; ++v) {
        const int k = kmap(v, half);
        vf.u[v] = pack2(vb[(size_t)(j0 + k) * rs + n],
                        vb[(size_t)(j0 + k + 1) * rs + n]);
      }
      o[nt] = wmma_bf16(pf, vf, o[nt]);
    }
    asm volatile("s_wait_dscnt 0" ::: "memory");
  }

  // ---- normalize + store bf16 for the AO GEMM ----
#pragma unroll
  for (int r = 0; r < 8; ++r) {
    const int i = i0 + r + 8 * half;
    const float inv = 1.0f / rsum[r];
    const size_t ro = (size_t)(b * L_ + i) * D_ + hh * HD_;
#pragma unroll
    for (int nt = 0; nt < 4; ++nt)
      outB[ro + nt * 16 + l16] = bfb(o[nt][r] * inv);
  }
}

// ---------------------------------------------------------------------------
// Launch: 2 modules x (embed + NL x [LN, QKV, attn, AO+res, LN, FC+gelu, PR+res])
// + final LNs + decode GEMM into d_out.
// Input order (setup_inputs dict): 0 ids, 1 s_mask, 2..19 m1_*, 20..37 m2_*,
// 38 m1_s_token.
// ---------------------------------------------------------------------------
extern "C" void kernel_launch(void* const* d_in, const int* in_sizes, int n_in,
                              void* d_out, int out_size, void* d_ws, size_t ws_size,
                              hipStream_t stream) {
  (void)in_sizes; (void)n_in; (void)out_size; (void)ws_size;

  const int*   ids   = (const int*)d_in[0];
  const int*   smask = (const int*)d_in[1];
  const float* stok  = (const float*)d_in[38];

  unsigned char* ws = (unsigned char*)d_ws;
  size_t off = 0;
  auto carve = [&](size_t bytes) -> void* {
    void* p = ws + off;
    off = (off + bytes + 255) & ~(size_t)255;
    return p;
  };
  float*          h      = (float*)carve((size_t)M_ * D_ * 4);
  float*          bufQKV = (float*)carve((size_t)M_ * 3 * D_ * 4);
  float*          h1f    = (float*)carve((size_t)M_ * D_ * 4);
  unsigned short* xbf    = (unsigned short*)carve((size_t)M_ * D_ * 2);
  unsigned short* attnbf = (unsigned short*)carve((size_t)M_ * D_ * 2);
  unsigned short* fcbf   = (unsigned short*)carve((size_t)M_ * 4 * D_ * 2);
  int*            lastS  = (int*)carve((size_t)M_ * 4);

  auto F = [&](int i) { return (const float*)d_in[i]; };

  lasts_kernel<<<1, 32, 0, stream>>>(smask, lastS);

  const dim3 gQKV(3 * D_ / 128, M_ / 128);
  const dim3 gD(D_ / 128, M_ / 128);
  const dim3 gFC(4 * D_ / 128, M_ / 128);
  const dim3 gDec(V_ / 128, M_ / 128);
  const dim3 gAttn(B_ * H_, L_ / 64);

  for (int mod = 0; mod < 2; ++mod) {
    const int base = 2 + mod * 18;
    const float* tok = F(base + 0);
    const float* pos = F(base + 1);
    if (mod == 0)
      embed1_kernel<<<M_, 256, 0, stream>>>(ids, smask, tok, stok, pos, h);
    else
      embed2_kernel<<<M_, 256, 0, stream>>>(ids, smask, tok, h1f, pos, h);

    for (int l = 0; l < NL_; ++l) {
      const float* ln1g = F(base + 2)  + (size_t)l * D_;
      const float* ln1b = F(base + 3)  + (size_t)l * D_;
      const float* qkvw = F(base + 4)  + (size_t)l * D_ * 3 * D_;
      const float* qkvb = F(base + 5)  + (size_t)l * 3 * D_;
      const float* aow  = F(base + 6)  + (size_t)l * D_ * D_;
      const float* aob  = F(base + 7)  + (size_t)l * D_;
      const float* ln2g = F(base + 8)  + (size_t)l * D_;
      const float* ln2b = F(base + 9)  + (size_t)l * D_;
      const float* fcw  = F(base + 10) + (size_t)l * D_ * 4 * D_;
      const float* fcb  = F(base + 11) + (size_t)l * 4 * D_;
      const float* prw  = F(base + 12) + (size_t)l * 4 * D_ * D_;
      const float* prb  = F(base + 13) + (size_t)l * D_;

      ln_kernel<true, false><<<M_, 256, 0, stream>>>(h, ln1g, ln1b, xbf, nullptr);
      gemm_kernel<false, false, false><<<gQKV, 256, 0, stream>>>(
          xbf, qkvw, qkvb, nullptr, bufQKV, nullptr, D_, 3 * D_);
      if (mod == 0)
        attn_kernel<false><<<gAttn, 128, 0, stream>>>(bufQKV, nullptr, attnbf);
      else
        attn_kernel<true><<<gAttn, 128, 0, stream>>>(bufQKV, lastS, attnbf);
      gemm_kernel<true, false, false><<<gD, 256, 0, stream>>>(
          attnbf, aow, aob, h, h, nullptr, D_, D_);
      ln_kernel<true, false><<<M_, 256, 0, stream>>>(h, ln2g, ln2b, xbf, nullptr);
      gemm_kernel<false, true, true><<<gFC, 256, 0, stream>>>(
          xbf, fcw, fcb, nullptr, nullptr, fcbf, D_, 4 * D_);
      gemm_kernel<true, false, false><<<gD, 256, 0, stream>>>(
          fcbf, prw, prb, h, h, nullptr, 4 * D_, D_);
    }

    const float* lnfg = F(base + 14);
    const float* lnfb = F(base + 15);
    if (mod == 0) {
      // module1 final LN kept in f32: feeds embed2 at s positions
      ln_kernel<false, true><<<M_, 256, 0, stream>>>(h, lnfg, lnfb, nullptr, h1f);
    } else {
      ln_kernel<true, false><<<M_, 256, 0, stream>>>(h, lnfg, lnfb, xbf, nullptr);
      const float* decw = F(base + 16);
      const float* decb = F(base + 17);
      gemm_kernel<false, false, false><<<gDec, 256, 0, stream>>>(
          xbf, decw, decb, nullptr, (float*)d_out, nullptr, D_, V_);
    }
  }
}
